// GeometricFrameAttention_24592982737135
// MI455X (gfx1250) — matrix operand order repaired and checked by reading, not compiled, not run on hardware
//
#include <hip/hip_runtime.h>
#include <hip/hip_bf16.h>

// ---------------- problem constants ----------------
#define NRES   768
#define C_S    384
#define C_Z    128
#define C_H    16
#define NH     12
#define PQ     4
#define PV     8
#define VPACKW 48      // 16 (v) + 24 (v_pts) + 8 pad
#define CATW   2112    // 192 + 288 + 96 + 1536
#define INF_F  1.0e9f
#define EPS_F  1e-8f
#define KSTEPS (NRES / 32)   // 24

typedef __attribute__((ext_vector_type(16))) __bf16 v16bf;
typedef __attribute__((ext_vector_type(8)))  float  v8f;
typedef __attribute__((ext_vector_type(4)))  unsigned int u32x4;
typedef __attribute__((ext_vector_type(8)))  unsigned int u32x8;

// ---------------- bf16 helpers (RNE via bit math) ---------------------------
__device__ inline unsigned short f2bfbits(float x) {
    unsigned int u = __float_as_uint(x);
    unsigned int r = (u + 0x7fffu + ((u >> 16) & 1u)) >> 16;
    return (unsigned short)r;
}
__device__ inline __bf16 f2bf(float x) {
    unsigned short us = f2bfbits(x);
    __bf16 b; __builtin_memcpy(&b, &us, 2); return b;
}
__device__ inline __bf16 ld_bf(const unsigned short* p) {
    unsigned short u = *p; __bf16 b; __builtin_memcpy(&b, &u, 2); return b;
}
__device__ inline void st_bf(unsigned short* p, float x) { *p = f2bfbits(x); }

// ---------------- gfx1250 async / TDM primitives (inline asm, portable) -----
// Per-lane async copy: 16B global -> LDS, tracked by ASYNCcnt.
__device__ inline void async_load_b128(unsigned lds_off, const void* gptr) {
    asm volatile("global_load_async_to_lds_b128 %0, %1, off"
                 :: "v"(lds_off), "v"((unsigned long long)gptr)
                 : "memory");
}
__device__ inline void wait_asynccnt0() {
    asm volatile("s_wait_asynccnt 0x0" ::: "memory");
}
// TDM: DMA a contiguous 2-D tile (rows x 128 f32, row stride 128 f32) to LDS.
// D# per CDNA5 ISA 8.3/8.4: g0 = {count=1, lds_addr, global_addr, type=2},
// g1 = {data_size=4B, tensor_dim0=128, tensor_dim1=rows, tile=128 x rows,
//       tensor_dim0_stride=128}.  Tracked by TENSORcnt.
__device__ inline void tdm_load_rows128(const float* gptr, unsigned lds_off,
                                        unsigned rows) {
    unsigned long long ga = (unsigned long long)gptr;
    u32x4 g0 = { 1u, lds_off, (unsigned)ga,
                 (unsigned)((ga >> 32) & 0x01FFFFFFull) | 0x80000000u };
    u32x8 g1 = { 2u << 16,            // data_size = 4B
                 128u << 16,          // tensor_dim0 = 128
                 rows << 16,          // tensor_dim1 = rows
                 128u << 16,          // tile_dim0 = 128
                 rows,                // tile_dim1 = rows, tile_dim2 = 0
                 128u,                // tensor_dim0_stride = 128
                 0u, 0u };
    asm volatile("tensor_load_to_lds %0, %1" :: "s"(g0), "s"(g1) : "memory");
}
__device__ inline unsigned lds_off_of(const void* p) {
    return (unsigned)(unsigned long long)p;   // flat LDS offset lives in addr[31:0]
}

// ---------------- WMMA fragment builders (CDNA5 wave32 layouts) -------------
// A 16x32 bf16: lane m = L%16, half = L/16; elem 2v+p holds K = (v>=4)*16 + half*8 + (v&3)*2 + p
template <typename F>
__device__ inline v16bf make_a_frag(F&& f, int lane) {
    v16bf a;
    const int m = lane & 15, half = lane >> 4;
#pragma unroll
    for (int v = 0; v < 8; ++v) {
#pragma unroll
        for (int p = 0; p < 2; ++p) {
            const int k = ((v >> 2) << 4) + (half << 3) + ((v & 3) << 1) + p;
            a[v * 2 + p] = f(m, k);
        }
    }
    return a;
}
// B 32x16 bf16: lane n = L%16, half = L/16; elem 2v+p holds K = (v>=4)*16 + half*8 + (v&3)*2 + p
template <typename F>
__device__ inline v16bf make_b_frag(F&& f, int lane) {
    v16bf b;
    const int n = lane & 15, half = lane >> 4;
#pragma unroll
    for (int v = 0; v < 8; ++v) {
#pragma unroll
        for (int p = 0; p < 2; ++p) {
            const int k = ((v >> 2) << 4) + (half << 3) + ((v & 3) << 1) + p;
            b[v * 2 + p] = f(k, n);
        }
    }
    return b;
}
// C/D 16x16 f32: elem r at (M = r + 8*(L/16), N = L%16)
__device__ inline v8f wmma_bf16(v16bf a, v16bf b, v8f c) {
    return __builtin_amdgcn_wmma_f32_16x16x32_bf16(
        false, a, false, b, (short)0, c, false, false);
}

// ---------------- K1: projections + frame apply + operand packing -----------
__global__ void k1_proj(const float* __restrict__ s, const float* __restrict__ rot,
                        const float* __restrict__ trans,
                        const float* __restrict__ Wq,  const float* __restrict__ bq,
                        const float* __restrict__ Wkv, const float* __restrict__ bkv,
                        const float* __restrict__ Wqp, const float* __restrict__ bqp,
                        const float* __restrict__ Wkvp,const float* __restrict__ bkvp,
                        const float* __restrict__ head_w,
                        unsigned short* __restrict__ Apack,
                        unsigned short* __restrict__ Bpack,
                        unsigned short* __restrict__ Vpack) {
    const int i = blockIdx.x;
    const int t = threadIdx.x;                 // 256 threads
    __shared__ float srow[C_S];
    __shared__ float proj[1152];
    __shared__ float hw_s[NH];
    for (int k = t; k < C_S; k += 256) srow[k] = s[i * C_S + k];
    if (t < NH) hw_s[t] = log1pf(__expf(head_w[t])) * sqrtf(2.0f / (27.0f * (float)PQ));
    __syncthreads();

    for (int f = t; f < 1152; f += 256) {
        const float* Wr; float acc;
        if (f < 192)      { Wr = Wq   + f * C_S;          acc = bq[f];          }
        else if (f < 576) { Wr = Wkv  + (f - 192) * C_S;  acc = bkv[f - 192];   }
        else if (f < 720) { Wr = Wqp  + (f - 576) * C_S;  acc = bqp[f - 576];   }
        else              { Wr = Wkvp + (f - 720) * C_S;  acc = bkvp[f - 720];  }
        for (int k = 0; k < C_S; ++k) acc += srow[k] * Wr[k];
        proj[f] = acc;
    }
    __syncthreads();

    if (t < NH) {
        const int h = t;
        float R[9], T[3];
#pragma unroll
        for (int u = 0; u < 9; ++u) R[u] = rot[i * 9 + u];
#pragma unroll
        for (int u = 0; u < 3; ++u) T[u] = trans[i * 3 + u];
        const float c1 = sqrtf(1.0f / (3.0f * (float)C_H));
        const float hw = hw_s[h];

        float qp[PQ][3]; float q2 = 0.f;
        for (int p = 0; p < PQ; ++p) {
            const float* raw = &proj[576 + (h * PQ + p) * 3];
            for (int c = 0; c < 3; ++c) {
                float r = R[c*3+0]*raw[0] + R[c*3+1]*raw[1] + R[c*3+2]*raw[2] + T[c];
                qp[p][c] = r; q2 += r * r;
            }
        }
        float kp[PQ][3], vp[PV][3]; float k2 = 0.f;
        for (int p = 0; p < PQ + PV; ++p) {
            const float* raw = &proj[720 + (h * (PQ + PV) + p) * 3];
            for (int c = 0; c < 3; ++c) {
                float r = R[c*3+0]*raw[0] + R[c*3+1]*raw[1] + R[c*3+2]*raw[2] + T[c];
                if (p < PQ) { kp[p][c] = r; k2 += r * r; } else { vp[p - PQ][c] = r; }
            }
        }
        // Apack: [c1*q(16) | hw*q_pts(12) | 1 | -0.5*hw*q2 | 0 0]
        unsigned short* Ar = Apack + (size_t)(h * NRES + i) * 32;
        for (int c = 0; c < 16; ++c) st_bf(&Ar[c], c1 * proj[h * C_H + c]);
        for (int p = 0; p < PQ; ++p)
            for (int c = 0; c < 3; ++c) st_bf(&Ar[16 + p * 3 + c], hw * qp[p][c]);
        st_bf(&Ar[28], 1.0f); st_bf(&Ar[29], -0.5f * hw * q2);
        st_bf(&Ar[30], 0.0f); st_bf(&Ar[31], 0.0f);
        // Bpack: [k(16) | k_pts(12) | -0.5*hw*k2 | 1 | 0 0]
        unsigned short* Br = Bpack + (size_t)(h * NRES + i) * 32;
        for (int c = 0; c < 16; ++c) st_bf(&Br[c], proj[192 + h * 2 * C_H + c]);
        for (int p = 0; p < PQ; ++p)
            for (int c = 0; c < 3; ++c) st_bf(&Br[16 + p * 3 + c], kp[p][c]);
        st_bf(&Br[28], -0.5f * hw * k2); st_bf(&Br[29], 1.0f);
        st_bf(&Br[30], 0.0f); st_bf(&Br[31], 0.0f);
        // Vpack: [v(16) | v_pts(24) | 0(8)]
        unsigned short* Vr = Vpack + (size_t)(h * NRES + i) * VPACKW;
        for (int c = 0; c < 16; ++c) st_bf(&Vr[c], proj[192 + h * 2 * C_H + 16 + c]);
        for (int p = 0; p < PV; ++p)
            for (int c = 0; c < 3; ++c) st_bf(&Vr[16 + p * 3 + c], vp[p][c]);
        for (int c = 40; c < VPACKW; ++c) st_bf(&Vr[c], 0.0f);
    }
}

// ---------------- K2: pre-bias logits, one WMMA per 16x16 tile --------------
__global__ void k2_logits(const unsigned short* __restrict__ Apack,
                          const unsigned short* __restrict__ Bpack,
                          const float* __restrict__ mask,
                          const float* __restrict__ bb,
                          float* __restrict__ aL) {
    const int it = blockIdx.x, jt = blockIdx.y, h = blockIdx.z;
    const int lane = threadIdx.x;
    const int i0 = it * 16, j0 = jt * 16;
    v16bf a = make_a_frag([&](int m, int k) {
        return ld_bf(Apack + (size_t)(h * NRES + i0 + m) * 32 + k); }, lane);
    v16bf b = make_b_frag([&](int k, int n) {
        return ld_bf(Bpack + (size_t)(h * NRES + j0 + n) * 32 + k); }, lane);
    v8f c = {};
    c = wmma_bf16(a, b, c);
    const float c2bb = 0.57735026919f * bb[h];
    const int n = lane & 15, mh = (lane >> 4) * 8;
    const float mj = mask[j0 + n];
#pragma unroll
    for (int r = 0; r < 8; ++r) {
        const int m = mh + r;
        const float v = c[r] + c2bb + INF_F * (mask[i0 + m] * mj - 1.0f);
        aL[((size_t)h * NRES + i0 + m) * NRES + j0 + n] = v;
    }
}

// ---------------- K3: pair bias from z (z pass 1) ---------------------------
// One wave per (j-tile, i). The 16x128 f32 z tile is one contiguous 8KB block:
// stream it with per-lane global_load_async_to_lds_b128, then WMMA from LDS.
__global__ void k3_bias(const float* __restrict__ z, const float* __restrict__ Wb,
                        float* __restrict__ aL) {
    const int jt = blockIdx.x, i = blockIdx.y;
    const int lane = threadIdx.x;          // 32
    const int j0 = jt * 16;
    __shared__ float zs[16][C_Z];          // 8KB
    const char* gbase = (const char*)(z + ((size_t)i * NRES + j0) * C_Z);
    const unsigned lbase = lds_off_of(&zs[0][0]);
#pragma unroll
    for (int u = 0; u < 16; ++u) {         // 512 x 16B chunks, 16 per lane
        const int chunk = u * 32 + lane;
        async_load_b128(lbase + chunk * 16, gbase + chunk * 16);
    }
    wait_asynccnt0();

    v8f c = {};
#pragma unroll
    for (int ks = 0; ks < 4; ++ks) {
        const int k0 = ks * 32;
        v16bf a = make_a_frag([&](int m, int k) {
            return f2bf(zs[m][k0 + k]); }, lane);
        v16bf b = make_b_frag([&](int k, int n) {
            return (n < NH) ? f2bf(Wb[n * C_Z + k0 + k]) : f2bf(0.0f); }, lane);
        c = wmma_bf16(a, b, c);
    }
    const float c2 = 0.57735026919f;
    const int n = lane & 15, mh = (lane >> 4) * 8;
    if (n < NH) {
#pragma unroll
        for (int r = 0; r < 8; ++r) {
            const int j = j0 + mh + r;
            aL[((size_t)n * NRES + i) * NRES + j] += c2 * c[r];
        }
    }
}

// ---------------- K4: row softmax + bf16 store ------------------------------
__global__ void k4_softmax(const float* __restrict__ aL,
                           unsigned short* __restrict__ a_bf) {
    const int i = blockIdx.x, h = blockIdx.y, t = threadIdx.x;   // 256 threads
    const float* row = aL + ((size_t)h * NRES + i) * NRES;
    __shared__ float red[256];
    float vals[3];
    float m = -INF_F;
#pragma unroll
    for (int u = 0; u < 3; ++u) { vals[u] = row[t + u * 256]; m = fmaxf(m, vals[u]); }
    red[t] = m; __syncthreads();
    for (int s2 = 128; s2 > 0; s2 >>= 1) {
        if (t < s2) red[t] = fmaxf(red[t], red[t + s2]);
        __syncthreads();
    }
    m = red[0]; __syncthreads();
    float sum = 0.f;
#pragma unroll
    for (int u = 0; u < 3; ++u) { vals[u] = __expf(vals[u] - m); sum += vals[u]; }
    red[t] = sum; __syncthreads();
    for (int s2 = 128; s2 > 0; s2 >>= 1) {
        if (t < s2) red[t] += red[t + s2];
        __syncthreads();
    }
    const float inv = 1.0f / red[0];
#pragma unroll
    for (int u = 0; u < 3; ++u)
        st_bf(a_bf + ((size_t)h * NRES + i) * NRES + t + u * 256, vals[u] * inv);
}

// ---------------- K5: o and o_pt (global frame): a @ Vpack, N=48 ------------
__global__ void k5_av(const unsigned short* __restrict__ a_bf,
                      const unsigned short* __restrict__ Vpack,
                      float* __restrict__ O1) {
    const int it = blockIdx.x, nt = blockIdx.y, h = blockIdx.z;
    const int lane = threadIdx.x;
    const int i0 = it * 16, n0 = nt * 16;
    v8f c = {};
    for (int ks = 0; ks < KSTEPS; ++ks) {
        const int k0 = ks * 32;
        v16bf a = make_a_frag([&](int m, int k) {
            return ld_bf(a_bf + ((size_t)h * NRES + i0 + m) * NRES + k0 + k); }, lane);
        v16bf b = make_b_frag([&](int k, int n) {
            return ld_bf(Vpack + (size_t)(h * NRES + k0 + k) * VPACKW + n0 + n); }, lane);
        c = wmma_bf16(a, b, c);
    }
    const int n = lane & 15, mh = (lane >> 4) * 8;
#pragma unroll
    for (int r = 0; r < 8; ++r)
        O1[((size_t)h * NRES + i0 + mh + r) * VPACKW + n0 + n] = c[r];
}

// ---------------- K5post: frame inverse, norms, scatter into cat ------------
__global__ void k5_post(const float* __restrict__ O1, const float* __restrict__ rot,
                        const float* __restrict__ trans, float* __restrict__ cat) {
    const int i = blockIdx.x, t = threadIdx.x;   // 128 threads
    __shared__ float R[9], T[3];
    if (t < 9) R[t] = rot[i * 9 + t];
    if (t < 3) T[t] = trans[i * 3 + t];
    __syncthreads();
    float* crow = cat + (size_t)i * CATW;
    for (int u = t; u < NH * C_H; u += 128) {          // o : 192
        const int h = u / C_H, c = u % C_H;
        crow[u] = O1[((size_t)h * NRES + i) * VPACKW + c];
    }
    for (int u = t; u < NH * PV; u += 128) {           // points: 96 (h,p)
        const int h = u / PV, p = u % PV;
        float g[3], l[3];
#pragma unroll
        for (int c = 0; c < 3; ++c)
            g[c] = O1[((size_t)h * NRES + i) * VPACKW + 16 + p * 3 + c] - T[c];
#pragma unroll
        for (int c = 0; c < 3; ++c)                    // R^T * g
            l[c] = R[0 * 3 + c] * g[0] + R[1 * 3 + c] * g[1] + R[2 * 3 + c] * g[2];
#pragma unroll
        for (int c = 0; c < 3; ++c)                    // o_pt_flat: 192 + c*96 + h*8 + p
            crow[192 + c * (NH * PV) + h * PV + p] = l[c];
        crow[480 + u] = sqrtf(l[0]*l[0] + l[1]*l[1] + l[2]*l[2] + EPS_F);
    }
}

// ---------------- K6: o_pair = a @ z (z pass 2) -----------------------------
// One 256-thread workgroup (8 waves) per residue i; wave w owns c-tile w.
// Double-buffered TDM pipeline: tensor_load_to_lds streams 32x128 f32 z tiles
// into LDS while 8 WMMAs consume the previous tile. Shared A-tile (12 heads x
// 32 K bf16 attention weights, rows 12..15 zero) is staged in LDS per K-step.
__global__ void __launch_bounds__(256) k6_opair(
        const unsigned short* __restrict__ a_bf,
        const float* __restrict__ z, float* __restrict__ cat) {
    const int i = blockIdx.x;
    const int t = threadIdx.x, wave = t >> 5, lane = t & 31;
    __shared__ float zs[2][32][C_Z];            // 2 x 16KB
    __shared__ unsigned short abuf[2][16][32];  // 2 x 1KB

    // zero the padded A rows (12..15) of both buffers
    for (int u = t; u < 2 * 4 * 32; u += 256) {
        const int b = u >> 7, r = (u >> 5) & 3, k = u & 31;
        abuf[b][12 + r][k] = 0;
    }
    // prologue: issue tile 0 DMA + stage A tile 0
    if (wave == 0 && lane == 0)
        tdm_load_rows128(z + (size_t)i * NRES * C_Z, lds_off_of(&zs[0][0][0]), 32u);
    if (t < 192) {
        const int h = t >> 4, kk = (t & 15) * 2;
        const unsigned int* p =
            (const unsigned int*)(a_bf + ((size_t)h * NRES + i) * NRES + kk);
        *(unsigned int*)&abuf[0][h][kk] = *p;
    }

    v8f acc = {};
    const int c0 = wave * 16;
    for (int ks = 0; ks < KSTEPS; ++ks) {
        const int cur = ks & 1;
        __builtin_amdgcn_s_wait_tensorcnt(0);   // tile `cur` DMA complete (wave 0)
        __syncthreads();                        // ...visible to all waves
        if (ks + 1 < KSTEPS) {
            const int k0n = (ks + 1) * 32;
            if (wave == 0 && lane == 0)
                tdm_load_rows128(z + ((size_t)i * NRES + k0n) * C_Z,
                                 lds_off_of(&zs[1 - cur][0][0]), 32u);
            if (t < 192) {
                const int h = t >> 4, kk = (t & 15) * 2;
                const unsigned int* p = (const unsigned int*)
                    (a_bf + ((size_t)h * NRES + i) * NRES + k0n + kk);
                *(unsigned int*)&abuf[1 - cur][h][kk] = *p;
            }
        }
        // compute on buffer `cur`
        v16bf a = make_a_frag([&](int m, int k) {
            return ld_bf(&abuf[cur][m][k]); }, lane);
        v16bf b = make_b_frag([&](int k, int n) {
            return f2bf(zs[cur][k][c0 + n]); }, lane);
        acc = wmma_bf16(a, b, acc);
        __syncthreads();                        // buffer reuse safety
    }
    const int n = lane & 15, mh = (lane >> 4) * 8;
#pragma unroll
    for (int r = 0; r < 8; ++r) {
        const int h = mh + r;
        if (h < NH)
            cat[(size_t)i * CATW + 576 + h * C_Z + c0 + n] = acc[r];
    }
}

// ---------------- K7: out = cat @ Wout^T + bout -----------------------------
__global__ void k7_out(const float* __restrict__ cat, const float* __restrict__ Wout,
                       const float* __restrict__ bout, float* __restrict__ out) {
    const int it = blockIdx.x, ot = blockIdx.y;
    const int lane = threadIdx.x;
    const int i0 = it * 16, o0 = ot * 16;
    v8f acc = {};
    for (int ks = 0; ks < CATW / 32; ++ks) {
        const int k0 = ks * 32;
        v16bf a = make_a_frag([&](int m, int k) {
            return f2bf(cat[(size_t)(i0 + m) * CATW + k0 + k]); }, lane);
        v16bf b = make_b_frag([&](int k, int n) {
            return f2bf(Wout[(size_t)(o0 + n) * CATW + k0 + k]); }, lane);
        acc = wmma_bf16(a, b, acc);
    }
    const int n = lane & 15, mh = (lane >> 4) * 8;
#pragma unroll
    for (int r = 0; r < 8; ++r)
        out[(size_t)(i0 + mh + r) * C_S + o0 + n] = acc[r] + bout[o0 + n];
}

// ---------------- launch ----------------------------------------------------
extern "C" void kernel_launch(void* const* d_in, const int* in_sizes, int n_in,
                              void* d_out, int out_size, void* d_ws, size_t ws_size,
                              hipStream_t stream) {
    const float* s      = (const float*)d_in[0];
    const float* z      = (const float*)d_in[1];
    const float* rot    = (const float*)d_in[2];
    const float* trans  = (const float*)d_in[3];
    const float* mask   = (const float*)d_in[4];
    const float* Wq     = (const float*)d_in[5];
    const float* bq     = (const float*)d_in[6];
    const float* Wkv    = (const float*)d_in[7];
    const float* bkv    = (const float*)d_in[8];
    const float* Wqp    = (const float*)d_in[9];
    const float* bqp    = (const float*)d_in[10];
    const float* Wkvp   = (const float*)d_in[11];
    const float* bkvp   = (const float*)d_in[12];
    const float* Wb     = (const float*)d_in[13];
    const float* bb     = (const float*)d_in[14];
    const float* head_w = (const float*)d_in[15];
    const float* Wout   = (const float*)d_in[16];
    const float* bout   = (const float*)d_in[17];
    float* out = (float*)d_out;

    // workspace layout (bytes, all 256B aligned)
    char* ws = (char*)d_ws;
    size_t off = 0;
    float*          aL    = (float*)(ws + off);          off += (size_t)NH * NRES * NRES * 4;  // 28.3 MB
    unsigned short* a_bf  = (unsigned short*)(ws + off); off += (size_t)NH * NRES * NRES * 2;  // 14.2 MB
    unsigned short* Apack = (unsigned short*)(ws + off); off += (size_t)NH * NRES * 32 * 2;
    unsigned short* Bpack = (unsigned short*)(ws + off); off += (size_t)NH * NRES * 32 * 2;
    unsigned short* Vpack = (unsigned short*)(ws + off); off += (size_t)NH * NRES * VPACKW * 2;
    float*          O1    = (float*)(ws + off);          off += (size_t)NH * NRES * VPACKW * 4;
    float*          cat   = (float*)(ws + off);          off += (size_t)NRES * CATW * 4;       // total ~52.8 MB

    k1_proj<<<NRES, 256, 0, stream>>>(s, rot, trans, Wq, bq, Wkv, bkv, Wqp, bqp,
                                      Wkvp, bkvp, head_w, Apack, Bpack, Vpack);
    k2_logits<<<dim3(NRES / 16, NRES / 16, NH), 32, 0, stream>>>(Apack, Bpack, mask, bb, aL);
    k3_bias<<<dim3(NRES / 16, NRES), 32, 0, stream>>>(z, Wb, aL);
    k4_softmax<<<dim3(NRES, NH), 256, 0, stream>>>(aL, a_bf);
    k5_av<<<dim3(NRES / 16, VPACKW / 16, NH), 32, 0, stream>>>(a_bf, Vpack, O1);
    k5_post<<<NRES, 128, 0, stream>>>(O1, rot, trans, cat);
    k6_opair<<<NRES, 256, 0, stream>>>(a_bf, z, cat);
    k7_out<<<dim3(NRES / 16, C_S / 16), 32, 0, stream>>>(cat, Wout, bout, out);
}
